// PCTConv_76373108457627
// MI455X (gfx1250) — compile-verified, matrix operand order checked by reading, not run on hardware
//
#include <hip/hip_runtime.h>
#include <hip/hip_bf16.h>
#include <math.h>

// Problem constants (match reference)
#define NNODES 20000
#define DIM    192
#define NHEADS 3
#define CO_    64
#define NEDGES 320000
#define NCELL  16
#define MMETA  48
#define NREL   4
#define EMETA  256
#define TSIZE  32
#define KNB    8
#define SSEM   128
#define TU_    100

#define KBLKS   6      // 192/32
#define COLTILE 12     // 192/16
#define BPACK_ELEMS (COLTILE * KBLKS * 32 * 16)   // 36864 f16 = 73728 B

typedef __attribute__((ext_vector_type(16))) _Float16 v16h;
typedef __attribute__((ext_vector_type(8)))  float    v8f;

__device__ __forceinline__ void atomicMaxFloat(float* addr, float val) {
    // mixed-sign float max via int/uint monotone bit patterns
    if (val >= 0.0f) atomicMax((int*)addr, __float_as_int(val));
    else             atomicMin((unsigned int*)addr, __float_as_uint(val));
}

__global__ void zero_kernel(float* p, int n) {
    int i = blockIdx.x * blockDim.x + threadIdx.x;
    if (i < n) p[i] = 0.0f;
}

__global__ void copy_kernel(float* dst, const float* __restrict__ src, int n) {
    int i = blockIdx.x * blockDim.x + threadIdx.x;
    if (i < n) dst[i] = src[i];
}

// Repack B[192,192] f32 row-major into f16 WMMA B-fragment order:
// Bp[((colTile*6 + kblk)*32 + lane)*16 + e], e=(j,pair) per ISA 16-bit B layout.
__global__ void pack_b_kernel(const float* __restrict__ B, _Float16* __restrict__ Bp) {
    int idx = blockIdx.x * blockDim.x + threadIdx.x;
    if (idx >= BPACK_ELEMS) return;
    int e    = idx & 15;
    int lane = (idx >> 4) & 31;
    int r2   = idx >> 9;
    int kblk = r2 % KBLKS;
    int colTile = r2 / KBLKS;
    int j = e >> 1, odd = e & 1;
    int kk  = ((j & 4) << 2) + (lane >> 4) * 8 + (j & 3) * 2 + odd;
    int K   = kblk * 32 + kk;
    int col = colTile * 16 + (lane & 15);
    Bp[idx] = (_Float16)B[(size_t)K * DIM + col];
}

// C[mrows,192] = A[mrows,192] x B[192,192]; A f32 (vectorized b128 loads +
// cvt_pk), B pre-packed f16 fragments (one 32B load each). f32 accum WMMA.
// One wave = one 16-row tile x four 16-col tiles (A fragment reused 4x).
__global__ void __launch_bounds__(256) gemm_wmma_f16(
    const float* __restrict__ A, const _Float16* __restrict__ Bpack,
    float* __restrict__ C, int mrows)
{
    int wg     = blockIdx.x * 8 + (threadIdx.x >> 5);   // global wave id
    int groups = (mrows >> 4) * 3;                      // rowTiles * (12/4) colGroups
    if (wg >= groups) return;                           // wave-uniform: EXEC stays full
    int rowTile  = wg / 3;
    int colGroup = wg % 3;
    int lane = threadIdx.x & 31;
    int half = lane >> 4;        // which 16-lane half
    int m16  = lane & 15;
    int row0 = rowTile << 4;
    int col0 = colGroup * 64;

    v8f acc[4] = {v8f{}, v8f{}, v8f{}, v8f{}};
    const float* arow = A + (size_t)(row0 + m16) * DIM;

    #pragma unroll
    for (int k0i = 0; k0i < KBLKS; ++k0i) {
        int k0 = k0i * 32;
        // A fragment: two contiguous 8-float runs per lane -> 4x b128 + cvt_pk
        const float4* ap0 = (const float4*)(arow + k0 + half * 8);
        const float4* ap1 = (const float4*)(arow + k0 + 16 + half * 8);
        float4 f0 = ap0[0], f1 = ap0[1], f2 = ap1[0], f3 = ap1[1];
        v16h a;
        a[0]  = (_Float16)f0.x; a[1]  = (_Float16)f0.y;
        a[2]  = (_Float16)f0.z; a[3]  = (_Float16)f0.w;
        a[4]  = (_Float16)f1.x; a[5]  = (_Float16)f1.y;
        a[6]  = (_Float16)f1.z; a[7]  = (_Float16)f1.w;
        a[8]  = (_Float16)f2.x; a[9]  = (_Float16)f2.y;
        a[10] = (_Float16)f2.z; a[11] = (_Float16)f2.w;
        a[12] = (_Float16)f3.x; a[13] = (_Float16)f3.y;
        a[14] = (_Float16)f3.z; a[15] = (_Float16)f3.w;
        #pragma unroll
        for (int t = 0; t < 4; ++t) {
            int colTile = colGroup * 4 + t;
            const v16h* bp = (const v16h*)(Bpack +
                ((size_t)((colTile * KBLKS + k0i) * 32 + lane) << 4));
            v16h b = *bp;   // 32B aligned fragment load
            acc[t] = __builtin_amdgcn_wmma_f32_16x16x32_f16(
                false, a, false, b, (short)0, acc[t], false, false);
        }
    }
    // C/D layout: VGPR i -> row = row0 + 8*half + i, col = lane%16
    #pragma unroll
    for (int t = 0; t < 4; ++t) {
        int col = col0 + t * 16 + m16;
        #pragma unroll
        for (int i = 0; i < 8; ++i)
            C[(size_t)(row0 + half * 8 + i) * DIM + col] = acc[t][i];
    }
}

// a_s/a_d per (node,head); also init emax=-inf, denom=0
__global__ void att_coef_kernel(
    const float* __restrict__ h, const float* __restrict__ att_s,
    const float* __restrict__ att_d, float* as_, float* ad_,
    float* emax, float* denom, int n)
{
    int idx = blockIdx.x * blockDim.x + threadIdx.x;
    if (idx >= n * NHEADS) return;
    int node = idx / NHEADS, hh = idx % NHEADS;
    const float* hr = h + (size_t)node * DIM + hh * CO_;
    const float* sv = att_s + hh * CO_;
    const float* dv = att_d + hh * CO_;
    float s1 = 0.f, s2 = 0.f;
    #pragma unroll 8
    for (int i = 0; i < CO_; ++i) { float v = hr[i]; s1 += v * sv[i]; s2 += v * dv[i]; }
    as_[idx] = s1; ad_[idx] = s2;
    emax[idx] = -1e30f; denom[idx] = 0.0f;
}

__global__ void edge_max_kernel(
    const int* __restrict__ src, const int* __restrict__ dst,
    const float* __restrict__ as_, const float* __restrict__ ad_,
    float* emax, int ne)
{
    int e = blockIdx.x * blockDim.x + threadIdx.x;
    if (e >= ne) return;
    int s = src[e], d = dst[e];
    #pragma unroll
    for (int hh = 0; hh < NHEADS; ++hh) {
        float x = as_[s * NHEADS + hh] + ad_[d * NHEADS + hh];
        float lr = x > 0.f ? x : 0.2f * x;
        atomicMaxFloat(&emax[d * NHEADS + hh], lr);
    }
}

// wave-per-edge: lanes cover the 192 components coalesced; atomics hit L2
__global__ void __launch_bounds__(256) edge_sum_kernel(
    const int* __restrict__ src, const int* __restrict__ dst,
    const float* __restrict__ as_, const float* __restrict__ ad_,
    const float* __restrict__ emax, const float* __restrict__ h,
    float* denom, float* outacc, int ne)
{
    int e = blockIdx.x * 8 + (threadIdx.x >> 5);
    if (e >= ne) return;
    int lane = threadIdx.x & 31;
    int s = src[e], d = dst[e];
    float ex[NHEADS];
    #pragma unroll
    for (int hh = 0; hh < NHEADS; ++hh) {
        float x = as_[s * NHEADS + hh] + ad_[d * NHEADS + hh];
        float lr = x > 0.f ? x : 0.2f * x;
        ex[hh] = __expf(lr - emax[d * NHEADS + hh]);
    }
    if (lane < NHEADS) atomicAdd(&denom[d * NHEADS + lane], ex[lane]);
    const float* hs = h + (size_t)s * DIM;
    float* od = outacc + (size_t)d * DIM;
    #pragma unroll
    for (int j = 0; j < 6; ++j) {
        int comp = j * 32 + lane;
        atomicAdd(&od[comp], ex[comp >> 6] * hs[comp]);
    }
}

// att = x + out/denom' + bias, reduce column mean into mx[c] (scaled partials)
__global__ void __launch_bounds__(192) finalize_mean_kernel(
    const float* __restrict__ x, const float* __restrict__ outacc,
    const float* __restrict__ denom, const float* __restrict__ bias,
    float* __restrict__ mx_row, int n)
{
    int dcol = threadIdx.x;
    int base = blockIdx.x * 64;
    int hh = dcol >> 6;
    float b = bias[dcol];
    float sum = 0.f;
    for (int r = 0; r < 64; ++r) {
        int node = base + r;
        if (node >= n) break;
        float den = denom[node * NHEADS + hh];
        den = den > 0.f ? den : 1.f;
        sum += x[(size_t)node * DIM + dcol] + outacc[(size_t)node * DIM + dcol] / den + b;
    }
    atomicAdd(&mx_row[dcol], sum * (1.0f / (float)n));
}

// Gauss-Seidel sweeps are per-column independent: thread = column
__global__ void __launch_bounds__(192) tissue_kernel(
    float* __restrict__ mx, const int* __restrict__ nb)
{
    int dcol = threadIdx.x;
    for (int sweep = 0; sweep < TU_; ++sweep)
        for (int t = 0; t < TSIZE; ++t) {
            float s = 0.f;
            #pragma unroll
            for (int k = 0; k < KNB; ++k) s += mx[nb[t * KNB + k] * DIM + dcol];
            mx[(NCELL + t) * DIM + dcol] = s * (1.0f / KNB);
        }
}

// full metagraph GAT for one relation in a single workgroup via LDS
__global__ void __launch_bounds__(256) meta_gat_kernel(
    const float* __restrict__ hM, const float* __restrict__ att_s,
    const float* __restrict__ att_d, const float* __restrict__ bias,
    const int* __restrict__ edges, float* __restrict__ outs, int r)
{
    __shared__ float as_[MMETA * NHEADS], ad_[MMETA * NHEADS];
    __shared__ float emax_[MMETA * NHEADS], den_[MMETA * NHEADS];
    __shared__ float oacc[MMETA * DIM];
    int tid = threadIdx.x;
    if (tid < MMETA * NHEADS) {
        int node = tid / NHEADS, hh = tid % NHEADS;
        const float* hr = hM + node * DIM + hh * CO_;
        float s1 = 0.f, s2 = 0.f;
        for (int i = 0; i < CO_; ++i) { float v = hr[i]; s1 += v * att_s[hh * CO_ + i]; s2 += v * att_d[hh * CO_ + i]; }
        as_[tid] = s1; ad_[tid] = s2; emax_[tid] = -1e30f; den_[tid] = 0.f;
    }
    for (int i = tid; i < MMETA * DIM; i += 256) oacc[i] = 0.f;
    __syncthreads();

    const int* src = edges;
    const int* dst = edges + EMETA;
    {   // pass 1: segment max (one edge per thread; blockDim == EMETA)
        int s = src[tid], d = dst[tid];
        #pragma unroll
        for (int hh = 0; hh < NHEADS; ++hh) {
            float x = as_[s * NHEADS + hh] + ad_[d * NHEADS + hh];
            float lr = x > 0.f ? x : 0.2f * x;
            atomicMaxFloat(&emax_[d * NHEADS + hh], lr);
        }
    }
    __syncthreads();
    {   // pass 2: exp, denom, weighted scatter
        int s = src[tid], d = dst[tid];
        float ex[NHEADS];
        #pragma unroll
        for (int hh = 0; hh < NHEADS; ++hh) {
            float x = as_[s * NHEADS + hh] + ad_[d * NHEADS + hh];
            float lr = x > 0.f ? x : 0.2f * x;
            ex[hh] = __expf(lr - emax_[d * NHEADS + hh]);
            atomicAdd(&den_[d * NHEADS + hh], ex[hh]);
        }
        for (int comp = 0; comp < DIM; ++comp)
            atomicAdd(&oacc[d * DIM + comp], ex[comp >> 6] * hM[s * DIM + comp]);
    }
    __syncthreads();
    for (int i = tid; i < MMETA * DIM; i += 256) {
        int node = i / DIM, dcol = i % DIM, hh = dcol >> 6;
        float den = den_[node * NHEADS + hh]; den = den > 0.f ? den : 1.f;
        float v = oacc[i] / den + bias[dcol];
        outs[(size_t)node * (NREL * DIM) + r * DIM + dcol] = fmaxf(v, 0.f); // relu
    }
}

// semantic attention fusion: one block per metagraph node
__global__ void __launch_bounds__(192) sem_att_kernel(
    const float* __restrict__ outs, const float* __restrict__ W,
    const float* __restrict__ b, const float* __restrict__ q,
    float* __restrict__ out)
{
    int m = blockIdx.x, tid = threadIdx.x;
    __shared__ float red[192];
    __shared__ float logits[NREL];
    __shared__ float beta[NREL];
    for (int r = 0; r < NREL; ++r) {
        float partial = 0.f;
        if (tid < SSEM) {
            float acc = b[tid];
            const float* o = outs + ((size_t)m * NREL + r) * DIM;
            for (int dd = 0; dd < DIM; ++dd) acc += o[dd] * W[dd * SSEM + tid];
            partial = tanhf(acc) * q[tid];
        }
        red[tid] = partial;
        __syncthreads();
        if (tid < 64) red[tid] += red[tid + 64] + red[tid + 128];
        __syncthreads();
        if (tid == 0) { float s = 0.f; for (int i = 0; i < 64; ++i) s += red[i]; logits[r] = s; }
        __syncthreads();
    }
    if (tid == 0) {
        float mxv = logits[0];
        for (int r = 1; r < NREL; ++r) mxv = fmaxf(mxv, logits[r]);
        float e[NREL], sum = 0.f;
        for (int r = 0; r < NREL; ++r) { e[r] = __expf(logits[r] - mxv); sum += e[r]; }
        for (int r = 0; r < NREL; ++r) beta[r] = e[r] / sum;
    }
    __syncthreads();
    float acc = 0.f;
    for (int r = 0; r < NREL; ++r) acc += outs[((size_t)m * NREL + r) * DIM + tid] * beta[r];
    out[(size_t)m * DIM + tid] = acc;
}

extern "C" void kernel_launch(void* const* d_in, const int* in_sizes, int n_in,
                              void* d_out, int out_size, void* d_ws, size_t ws_size,
                              hipStream_t stream) {
    const float* ppi_x        = (const float*)d_in[0];
    const float* metagraph_x  = (const float*)d_in[1];
    const float* ppi_lin      = (const float*)d_in[2];
    const float* ppi_att_src  = (const float*)d_in[3];
    const float* ppi_att_dst  = (const float*)d_in[4];
    const float* ppi_bias     = (const float*)d_in[5];
    const float* meta_lin     = (const float*)d_in[6];
    const float* meta_att_src = (const float*)d_in[7];
    const float* meta_att_dst = (const float*)d_in[8];
    const float* meta_bias    = (const float*)d_in[9];
    const float* W_sem        = (const float*)d_in[10];
    const float* b_sem        = (const float*)d_in[11];
    const float* q_sem        = (const float*)d_in[12];
    const int*   ppi_edges    = (const int*)d_in[13];
    const int*   meta_edges   = (const int*)d_in[14];
    const int*   tissue_nb    = (const int*)d_in[15];

    float* ws = (float*)d_ws;
    size_t off = 0;
    float* h      = ws + off; off += (size_t)NNODES * DIM;   // 15.36 MB
    float* outacc = ws + off; off += (size_t)NNODES * DIM;   // 15.36 MB
    float* as_    = ws + off; off += (size_t)NNODES * NHEADS;
    float* ad_    = ws + off; off += (size_t)NNODES * NHEADS;
    float* emax   = ws + off; off += (size_t)NNODES * NHEADS;
    float* denom  = ws + off; off += (size_t)NNODES * NHEADS;
    float* mx     = ws + off; off += (size_t)MMETA * DIM;
    float* hmeta  = ws + off; off += (size_t)MMETA * DIM;
    float* outs   = ws + off; off += (size_t)MMETA * NREL * DIM;
    _Float16* bpack = (_Float16*)(ws + off); off += BPACK_ELEMS / 2;  // 72 KB, 32B aligned

    // mx = metagraph_x (working copy; inputs never mutated)
    copy_kernel<<<(MMETA * DIM + 255) / 256, 256, 0, stream>>>(mx, metagraph_x, MMETA * DIM);

    const int gemm_groups_ppi = (NNODES / 16) * 3;            // 3750 wave tile-groups
    const int gemm_blocks_ppi = (gemm_groups_ppi + 7) / 8;
    const int pack_blocks = (BPACK_ELEMS + 255) / 256;
    for (int c = 0; c < NCELL; ++c) {
        pack_b_kernel<<<pack_blocks, 256, 0, stream>>>(
            ppi_lin + (size_t)c * DIM * DIM, bpack);
        gemm_wmma_f16<<<gemm_blocks_ppi, 256, 0, stream>>>(
            ppi_x + (size_t)c * NNODES * DIM, bpack, h, NNODES);
        zero_kernel<<<(NNODES * DIM + 255) / 256, 256, 0, stream>>>(outacc, NNODES * DIM);
        att_coef_kernel<<<(NNODES * NHEADS + 255) / 256, 256, 0, stream>>>(
            h, ppi_att_src + (size_t)c * DIM, ppi_att_dst + (size_t)c * DIM,
            as_, ad_, emax, denom, NNODES);
        const int* esrc = ppi_edges + (size_t)c * 2 * NEDGES;
        edge_max_kernel<<<(NEDGES + 255) / 256, 256, 0, stream>>>(
            esrc, esrc + NEDGES, as_, ad_, emax, NEDGES);
        edge_sum_kernel<<<(NEDGES + 7) / 8, 256, 0, stream>>>(
            esrc, esrc + NEDGES, as_, ad_, emax, h, denom, outacc, NEDGES);
        finalize_mean_kernel<<<(NNODES + 63) / 64, 192, 0, stream>>>(
            ppi_x + (size_t)c * NNODES * DIM, outacc, denom,
            ppi_bias + (size_t)c * DIM, mx + (size_t)c * DIM, NNODES);
    }

    tissue_kernel<<<1, 192, 0, stream>>>(mx, tissue_nb);

    const int gemm_groups_meta = (MMETA / 16) * 3;            // 9 tile-groups
    for (int r = 0; r < NREL; ++r) {
        pack_b_kernel<<<pack_blocks, 256, 0, stream>>>(
            meta_lin + (size_t)r * DIM * DIM, bpack);
        gemm_wmma_f16<<<(gemm_groups_meta + 7) / 8, 256, 0, stream>>>(
            mx, bpack, hmeta, MMETA);
        meta_gat_kernel<<<1, 256, 0, stream>>>(
            hmeta, meta_att_src + (size_t)r * DIM, meta_att_dst + (size_t)r * DIM,
            meta_bias + (size_t)r * DIM, meta_edges + (size_t)r * 2 * EMETA, outs, r);
    }

    sem_att_kernel<<<MMETA, 192, 0, stream>>>(outs, W_sem, b_sem, q_sem, (float*)d_out);
}
